// EncoderLayer_78125455114454
// MI455X (gfx1250) — matrix operand order repaired and checked by reading, not compiled, not run on hardware
//
#include <hip/hip_runtime.h>
#include <hip/hip_bf16.h>
#include <math.h>

typedef __bf16 bf16_t;
typedef __bf16 v16bf __attribute__((ext_vector_type(16)));
typedef __bf16 v8bf  __attribute__((ext_vector_type(8)));
typedef float  v8f   __attribute__((ext_vector_type(8)));

#define S_LEN  2048
#define BATCH  2
#define DMODEL 1024
#define NHEADS 16
#define DEPTH  64
#define HIDDEN 4096
#define MROWS  (BATCH * S_LEN)   // 4096 total rows

// ---- fragment loaders ---------------------------------------------------
// A-fragment (16x32 bf16): lane m = L&15, khalf = L>>4 holds
//   K = khalf*8 + {0..7} and K = 16 + khalf*8 + {0..7}   (two 16B chunks)
__device__ inline v16bf ldA(const bf16_t* p) {
  v8bf lo = *(const v8bf*)p;
  v8bf hi = *(const v8bf*)(p + 16);
  v16bf a;
#pragma unroll
  for (int i = 0; i < 8; ++i) { a[i] = lo[i]; a[i + 8] = hi[i]; }
  return a;
}

__device__ inline v8f wmma_bf16(v16bf a, v16bf b, v8f c) {
  // (neg_a, A, neg_b, B, c_mod, C, reuse_a, reuse_b)
  return __builtin_amdgcn_wmma_f32_16x16x32_bf16(false, a, false, b, (short)0, c,
                                                 false, false);
}

// ---- cast f32 -> bf16 ---------------------------------------------------
__global__ __launch_bounds__(256) void cast_bf16_kernel(const float* __restrict__ in,
                                                        bf16_t* __restrict__ out, int n) {
  int i = blockIdx.x * 256 + threadIdx.x;
  if (i < n) out[i] = (bf16_t)in[i];
}

// ---- W[K,N] f32 -> Wt[N,K] bf16 (LDS tiled transpose) -------------------
__global__ __launch_bounds__(256) void transpose_bf16_kernel(const float* __restrict__ W,
                                                             bf16_t* __restrict__ Wt,
                                                             int K, int N) {
  __shared__ float t[32][33];
  const int n0 = blockIdx.x * 32, k0 = blockIdx.y * 32;
  const int tx = threadIdx.x, ty = threadIdx.y;   // 32 x 8
#pragma unroll
  for (int i = 0; i < 4; ++i)
    t[ty * 4 + i][tx] = W[(size_t)(k0 + ty * 4 + i) * N + n0 + tx];
  __syncthreads();
#pragma unroll
  for (int i = 0; i < 4; ++i)
    Wt[(size_t)(n0 + ty * 4 + i) * K + k0 + tx] = (bf16_t)t[tx][ty * 4 + i];
}

// ---- GEMM: C[M,N] = A[M,K](bf16) * Bt[N,K]^T(bf16) + bias ---------------
// 256 threads = 8 waves; block tile 128x64; each wave a 32x32 tile
// (2 A-frags x 2 B-frags -> 4 WMMA per 32-K step; 32 B/lane per WMMA).
enum { EP_BF16 = 0, EP_VT = 1, EP_F32 = 2, EP_RELU_BF16 = 3 };

template <int EP>
__global__ __launch_bounds__(256) void gemm_bf16(const bf16_t* __restrict__ A,
                                                 const bf16_t* __restrict__ Bt,
                                                 const float* __restrict__ bias,
                                                 float* __restrict__ outF,
                                                 bf16_t* __restrict__ outB,
                                                 int M, int N, int K) {
  const int lane = threadIdx.x & 31;
  const int wid  = threadIdx.x >> 5;
  const int ln = lane & 15;
  const int kh = lane >> 4;                       // K-half for A/B, M-half for C
  const int m0 = blockIdx.y * 128 + (wid & 3) * 32;
  const int n0 = blockIdx.x * 64 + (wid >> 2) * 32;

  const bf16_t* a0r = A  + (size_t)(m0 + ln) * K;
  const bf16_t* a1r = A  + (size_t)(m0 + 16 + ln) * K;
  const bf16_t* b0r = Bt + (size_t)(n0 + ln) * K;
  const bf16_t* b1r = Bt + (size_t)(n0 + 16 + ln) * K;

  v8f c00 = {}, c01 = {}, c10 = {}, c11 = {};
  for (int k0 = 0; k0 < K; k0 += 64) {
    // prefetch streamed activation rows one cacheline-pair ahead
    __builtin_prefetch(a0r + k0 + 128, 0, 3);
    __builtin_prefetch(a1r + k0 + 128, 0, 3);
#pragma unroll
    for (int ks = 0; ks < 2; ++ks) {
      const int kk = k0 + ks * 32;
      v16bf aA = ldA(a0r + kk + kh * 8);
      v16bf aB = ldA(a1r + kk + kh * 8);
      v16bf b0 = ldA(b0r + kk + kh * 8);   // same K order as A (two 16B chunks)
      v16bf b1 = ldA(b1r + kk + kh * 8);
      c00 = wmma_bf16(aA, b0, c00);
      c01 = wmma_bf16(aA, b1, c01);
      c10 = wmma_bf16(aB, b0, c10);
      c11 = wmma_bf16(aB, b1, c11);
    }
  }

  v8f cf[2][2] = {{c00, c01}, {c10, c11}};
#pragma unroll
  for (int ai = 0; ai < 2; ++ai) {
#pragma unroll
    for (int bi = 0; bi < 2; ++bi) {
      const int n = n0 + bi * 16 + ln;
      const float bs = bias[n];
#pragma unroll
      for (int j = 0; j < 8; ++j) {
        const int m = m0 + ai * 16 + kh * 8 + j;        // C layout: M = 8*half + vgpr
        float v = cf[ai][bi][j] + bs;
        if (EP == EP_F32) {
          outF[(size_t)m * N + n] = v;
        } else if (EP == EP_BF16) {
          outB[(size_t)m * N + n] = (bf16_t)v;
        } else if (EP == EP_RELU_BF16) {
          outB[(size_t)m * N + n] = (bf16_t)fmaxf(v, 0.f);
        } else {  // EP_VT: scatter V projection to vt[B,H,D,S]
          const int b = m >> 11, s = m & (S_LEN - 1);
          const int h = n >> 6,  d = n & 63;
          outB[(((size_t)b * NHEADS + h) * DEPTH + d) * S_LEN + s] = (bf16_t)v;
        }
      }
    }
  }
}

// ---- fused flash attention ----------------------------------------------
// grid: (S/64, B*H), block 128 (4 waves); each wave: 16 query rows x full S keys
__global__ __launch_bounds__(128) void attn_kernel(const bf16_t* __restrict__ q,
                                                   const bf16_t* __restrict__ k,
                                                   const bf16_t* __restrict__ vt,
                                                   const int* __restrict__ mask,
                                                   bf16_t* __restrict__ ctx) {
  __shared__ __align__(64) bf16_t pbuf[4][16 * 32];

  const int lane = threadIdx.x & 31;
  const int wid  = threadIdx.x >> 5;
  const int ln = lane & 15, kh = lane >> 4;
  const int q0 = blockIdx.x * 64 + wid * 16;
  const int b  = blockIdx.y >> 4;
  const int h  = blockIdx.y & 15;

  // Q fragments for K-dim 0..31 and 32..63 (held for whole key loop)
  const bf16_t* qrow = q + ((size_t)(b * S_LEN + q0 + ln) * DMODEL + h * DEPTH);
  const v16bf aq0 = ldA(qrow + kh * 8);
  const v16bf aq1 = ldA(qrow + 32 + kh * 8);

  float rowmax[8], rowsum[8];
#pragma unroll
  for (int j = 0; j < 8; ++j) { rowmax[j] = -3.0e38f; rowsum[j] = 0.f; }
  v8f acc[4] = {{}, {}, {}, {}};

  const int* mrow = mask + b * S_LEN;
  bf16_t* myp = pbuf[wid];

  for (int kt = 0; kt < S_LEN; kt += 32) {
    // --- scores for a 32-key tile (two 16-wide N tiles) ---
    v8f sc[2];
#pragma unroll
    for (int t = 0; t < 2; ++t) {
      const int key = kt + t * 16 + ln;
      const bf16_t* krow = k + ((size_t)(b * S_LEN + key) * DMODEL + h * DEPTH);
      v16bf kb0 = ldA(krow + kh * 8);
      v16bf kb1 = ldA(krow + 32 + kh * 8);
      v8f c = {};
      c = wmma_bf16(aq0, kb0, c);
      c = wmma_bf16(aq1, kb1, c);
      const float madd = (float)mrow[key] * -1e9f;
#pragma unroll
      for (int j = 0; j < 8; ++j) c[j] = c[j] * 0.125f + madd;  // 1/sqrt(64)
      sc[t] = c;
    }

    // --- online softmax (row reductions across 16-lane half-waves) ---
#pragma unroll
    for (int j = 0; j < 8; ++j) {
      float lmax = fmaxf(sc[0][j], sc[1][j]);
#pragma unroll
      for (int d = 1; d < 16; d <<= 1) lmax = fmaxf(lmax, __shfl_xor(lmax, d, 32));
      const float nm   = fmaxf(rowmax[j], lmax);
      const float corr = __expf(rowmax[j] - nm);
      rowmax[j] = nm;
      const float p0 = __expf(sc[0][j] - nm);
      const float p1 = __expf(sc[1][j] - nm);
      const int row = kh * 8 + j;                   // C layout row within tile
      myp[row * 32 + ln]      = (bf16_t)p0;
      myp[row * 32 + 16 + ln] = (bf16_t)p1;
      float ps = p0 + p1;
#pragma unroll
      for (int d = 1; d < 16; d <<= 1) ps += __shfl_xor(ps, d, 32);
      rowsum[j] = rowsum[j] * corr + ps;
#pragma unroll
      for (int t = 0; t < 4; ++t) acc[t][j] *= corr;
    }

    // --- P (via LDS, re-shaped to A layout) times V tile ---
    const v16bf pa = ldA(myp + ln * 32 + kh * 8);
#pragma unroll
    for (int t = 0; t < 4; ++t) {
      const int d = t * 16 + ln;
      const bf16_t* vrow =
          vt + (((size_t)b * NHEADS + h) * DEPTH + d) * S_LEN + kt + kh * 8;
      v16bf bvv = ldA(vrow);
      acc[t] = wmma_bf16(pa, bvv, acc[t]);
    }
  }

  // --- normalize and write context [B,S,D] bf16 ---
#pragma unroll
  for (int t = 0; t < 4; ++t)
#pragma unroll
    for (int j = 0; j < 8; ++j) {
      const int qr = q0 + kh * 8 + j;
      const float v = acc[t][j] / rowsum[j];
      ctx[((size_t)(b * S_LEN) + qr) * DMODEL + h * DEPTH + t * 16 + ln] = (bf16_t)v;
    }
}

// ---- residual + LayerNorm (ddof=1, alpha*(x-mean)/(std+eps)+beta) -------
template <bool WB>
__global__ __launch_bounds__(256) void ln_kernel(const float* __restrict__ xa,
                                                 const float* __restrict__ xb,
                                                 const float* __restrict__ alpha,
                                                 const float* __restrict__ beta,
                                                 float* __restrict__ outF,
                                                 bf16_t* __restrict__ outB) {
  __shared__ float red[256];
  const int row = blockIdx.x;
  const int tid = threadIdx.x;
  const float* pa = xa + (size_t)row * DMODEL;
  const float* pb = xb + (size_t)row * DMODEL;
  float v[4];
  float s = 0.f;
#pragma unroll
  for (int i = 0; i < 4; ++i) { v[i] = pa[i * 256 + tid] + pb[i * 256 + tid]; s += v[i]; }
  red[tid] = s; __syncthreads();
  for (int o = 128; o > 0; o >>= 1) { if (tid < o) red[tid] += red[tid + o]; __syncthreads(); }
  const float mean = red[0] / DMODEL;
  __syncthreads();
  float ss = 0.f;
#pragma unroll
  for (int i = 0; i < 4; ++i) { const float d = v[i] - mean; ss += d * d; }
  red[tid] = ss; __syncthreads();
  for (int o = 128; o > 0; o >>= 1) { if (tid < o) red[tid] += red[tid + o]; __syncthreads(); }
  const float stdv = sqrtf(red[0] / (float)(DMODEL - 1));
  const float inv  = 1.f / (stdv + 1e-6f);
#pragma unroll
  for (int i = 0; i < 4; ++i) {
    const int c = i * 256 + tid;
    const float o = alpha[c] * (v[i] - mean) * inv + beta[c];
    outF[(size_t)row * DMODEL + c] = o;
    if (WB) outB[(size_t)row * DMODEL + c] = (bf16_t)o;
  }
}

// ---- launcher -----------------------------------------------------------
extern "C" void kernel_launch(void* const* d_in, const int* in_sizes, int n_in,
                              void* d_out, int out_size, void* d_ws, size_t ws_size,
                              hipStream_t stream) {
  (void)in_sizes; (void)n_in; (void)out_size; (void)ws_size;
  const float* x      = (const float*)d_in[0];
  const int*   mask   = (const int*)d_in[1];
  const float* Wq     = (const float*)d_in[2];
  const float* bq     = (const float*)d_in[3];
  const float* Wk     = (const float*)d_in[4];
  const float* bk     = (const float*)d_in[5];
  const float* Wv     = (const float*)d_in[6];
  const float* bv     = (const float*)d_in[7];
  const float* Wo     = (const float*)d_in[8];
  const float* bo     = (const float*)d_in[9];
  const float* alpha1 = (const float*)d_in[10];
  const float* beta1  = (const float*)d_in[11];
  const float* W1     = (const float*)d_in[12];
  const float* b1     = (const float*)d_in[13];
  const float* W2     = (const float*)d_in[14];
  const float* b2     = (const float*)d_in[15];
  const float* alpha2 = (const float*)d_in[16];
  const float* beta2  = (const float*)d_in[17];
  float* out = (float*)d_out;

  char* ws = (char*)d_ws;
  size_t off = 0;
  auto alloc = [&](size_t bytes) -> char* {
    char* p = ws + off;
    off = (off + bytes + 511) & ~(size_t)511;
    return p;
  };
  bf16_t* xbf   = (bf16_t*)alloc((size_t)MROWS * DMODEL * 2);
  bf16_t* Wqt   = (bf16_t*)alloc((size_t)DMODEL * DMODEL * 2);
  bf16_t* Wkt   = (bf16_t*)alloc((size_t)DMODEL * DMODEL * 2);
  bf16_t* Wvt   = (bf16_t*)alloc((size_t)DMODEL * DMODEL * 2);
  bf16_t* Wot   = (bf16_t*)alloc((size_t)DMODEL * DMODEL * 2);
  bf16_t* W1t   = (bf16_t*)alloc((size_t)HIDDEN * DMODEL * 2);
  bf16_t* W2t   = (bf16_t*)alloc((size_t)DMODEL * HIDDEN * 2);
  bf16_t* qbf   = (bf16_t*)alloc((size_t)MROWS * DMODEL * 2);
  bf16_t* kbf   = (bf16_t*)alloc((size_t)MROWS * DMODEL * 2);
  bf16_t* vtb   = (bf16_t*)alloc((size_t)MROWS * DMODEL * 2);   // [B,H,D,S]
  bf16_t* ctx   = (bf16_t*)alloc((size_t)MROWS * DMODEL * 2);
  float*  y     = (float*)alloc((size_t)MROWS * DMODEL * 4);    // attn-proj out, reused for ffn2
  float*  out1f = (float*)alloc((size_t)MROWS * DMODEL * 4);
  bf16_t* out1b = (bf16_t*)alloc((size_t)MROWS * DMODEL * 2);
  bf16_t* hid   = (bf16_t*)alloc((size_t)MROWS * HIDDEN * 2);

  // 1. activations -> bf16
  cast_bf16_kernel<<<(MROWS * DMODEL) / 256, 256, 0, stream>>>(x, xbf, MROWS * DMODEL);
  // 2. weights -> transposed bf16
  transpose_bf16_kernel<<<dim3(DMODEL / 32, DMODEL / 32), dim3(32, 8), 0, stream>>>(Wq, Wqt, DMODEL, DMODEL);
  transpose_bf16_kernel<<<dim3(DMODEL / 32, DMODEL / 32), dim3(32, 8), 0, stream>>>(Wk, Wkt, DMODEL, DMODEL);
  transpose_bf16_kernel<<<dim3(DMODEL / 32, DMODEL / 32), dim3(32, 8), 0, stream>>>(Wv, Wvt, DMODEL, DMODEL);
  transpose_bf16_kernel<<<dim3(DMODEL / 32, DMODEL / 32), dim3(32, 8), 0, stream>>>(Wo, Wot, DMODEL, DMODEL);
  transpose_bf16_kernel<<<dim3(HIDDEN / 32, DMODEL / 32), dim3(32, 8), 0, stream>>>(W1, W1t, DMODEL, HIDDEN);
  transpose_bf16_kernel<<<dim3(DMODEL / 32, HIDDEN / 32), dim3(32, 8), 0, stream>>>(W2, W2t, HIDDEN, DMODEL);

  const dim3 gproj(DMODEL / 64, MROWS / 128);
  // 3-5. Q/K/V projections (V written transposed [B,H,D,S])
  gemm_bf16<EP_BF16><<<gproj, 256, 0, stream>>>(xbf, Wqt, bq, nullptr, qbf, MROWS, DMODEL, DMODEL);
  gemm_bf16<EP_BF16><<<gproj, 256, 0, stream>>>(xbf, Wkt, bk, nullptr, kbf, MROWS, DMODEL, DMODEL);
  gemm_bf16<EP_VT>  <<<gproj, 256, 0, stream>>>(xbf, Wvt, bv, nullptr, vtb, MROWS, DMODEL, DMODEL);
  // 6. fused masked-softmax attention
  attn_kernel<<<dim3(S_LEN / 64, BATCH * NHEADS), 128, 0, stream>>>(qbf, kbf, vtb, mask, ctx);
  // 7. output projection (f32)
  gemm_bf16<EP_F32><<<gproj, 256, 0, stream>>>(ctx, Wot, bo, y, nullptr, MROWS, DMODEL, DMODEL);
  // 8. residual + LN1 (f32 + bf16 copy for FFN)
  ln_kernel<true><<<MROWS, 256, 0, stream>>>(x, y, alpha1, beta1, out1f, out1b);
  // 9. FFN up + ReLU (bf16)
  gemm_bf16<EP_RELU_BF16><<<dim3(HIDDEN / 64, MROWS / 128), 256, 0, stream>>>(out1b, W1t, b1, nullptr, hid, MROWS, HIDDEN, DMODEL);
  // 10. FFN down (f32, reuse y)
  gemm_bf16<EP_F32><<<gproj, 256, 0, stream>>>(hid, W2t, b2, y, nullptr, MROWS, DMODEL, HIDDEN);
  // 11. residual + LN2 -> d_out
  ln_kernel<false><<<MROWS, 256, 0, stream>>>(out1f, y, alpha2, beta2, out, nullptr);
}
